// MultiHeadLatentAttention_17617955848834
// MI455X (gfx1250) — compile-verified, matrix operand order checked by reading, not run on hardware
//
#include <hip/hip_runtime.h>
#include <hip/hip_bf16.h>
#include <math.h>

// ---------------------------------------------------------------------------
// Types for CDNA5 WMMA (gfx1250, wave32)
// ---------------------------------------------------------------------------
typedef __bf16 v16bf __attribute__((ext_vector_type(16)));
typedef float  v8f   __attribute__((ext_vector_type(8)));
typedef int    v4i   __attribute__((vector_size(16)));

struct __align__(16) U128 { unsigned int x, y, z, w; };
union Frag { v16bf v; U128 q[2]; };

__device__ __forceinline__ __bf16 f2bf(float f) {
  union { float f; unsigned u; } a; a.f = f;
  unsigned u = a.u;
  u += 0x7FFFu + ((u >> 16) & 1u);            // round-to-nearest-even
  union { unsigned short s; __bf16 b; } r;
  r.s = (unsigned short)(u >> 16);
  return r.b;
}

// Problem constants
#define DIMC   2048
#define HC     16
#define SC     2048
#define BC     2
#define QLORA  1536
#define KVLORA 512
#define MROWS  (BC * SC)          // 4096

// ---------------------------------------------------------------------------
// gfx1250 async global->LDS copy (ASYNCcnt path), guarded with fallback
// ---------------------------------------------------------------------------
#ifndef __has_builtin
#define __has_builtin(x) 0
#endif
#if __has_builtin(__builtin_amdgcn_global_load_async_to_lds_b128)
#define HAVE_ASYNC_LDS 1
#else
#define HAVE_ASYNC_LDS 0
#endif
#if __has_builtin(__builtin_amdgcn_s_wait_asynccnt)
#define HAVE_WAIT_ASYNC 1
#else
#define HAVE_WAIT_ASYNC 0
#endif

__device__ __forceinline__ void g2l16(const __bf16* g, U128* l) {
#if HAVE_ASYNC_LDS
  __builtin_amdgcn_global_load_async_to_lds_b128(
      (__attribute__((address_space(1))) v4i*)(uintptr_t)g,
      (__attribute__((address_space(3))) v4i*)(unsigned)(uintptr_t)l,
      0, 0);
#else
  *l = *(const U128*)g;
#endif
}

__device__ __forceinline__ void wait_async() {
#if HAVE_ASYNC_LDS
#if HAVE_WAIT_ASYNC
  __builtin_amdgcn_s_wait_asynccnt(0);
#else
  asm volatile("s_wait_asynccnt 0x0" ::: "memory");
#endif
#endif
}

// ---------------------------------------------------------------------------
// f32 -> bf16 conversion (row-major, for activations)
// ---------------------------------------------------------------------------
__global__ __launch_bounds__(256) void cvt_bf16_kernel(const float* __restrict__ in,
                                                       __bf16* __restrict__ out, int n) {
  int i = blockIdx.x * 256 + threadIdx.x;
  if (i < n) out[i] = f2bf(in[i]);
}

// ---------------------------------------------------------------------------
// f32 (K x N) -> bf16 transposed (N x K).  32x32 tiles, 32x8 threads.
// Removes all hot-loop LDS transposition from the GEMMs.
// ---------------------------------------------------------------------------
__global__ __launch_bounds__(256) void cvtT_bf16_kernel(const float* __restrict__ in,
                                                        __bf16* __restrict__ out,
                                                        int K, int N) {
  __shared__ __bf16 t[32][33];
  int n0 = blockIdx.x * 32, k0 = blockIdx.y * 32;
  int tx = threadIdx.x, ty = threadIdx.y;        // 32 x 8
  for (int i = ty; i < 32; i += 8)
    t[i][tx] = f2bf(in[(size_t)(k0 + i) * N + n0 + tx]);
  __syncthreads();
  for (int i = ty; i < 32; i += 8)
    out[(size_t)(n0 + i) * K + k0 + tx] = t[tx][i];
}

// ---------------------------------------------------------------------------
// GEMM: C(MxN,f32) = A(MxK,bf16 row-major) @ BT(NxK,bf16, i.e. B transposed)
// Block = 256 threads = 8 waves, tile 128(M) x 128(N), K-step 32,
// double-buffered LDS fed by async global->LDS copies.
// Wave (wm,wn): wm in {0..3} x 32 rows, wn in {0,1} x 64 cols -> 8 wmma/step.
// ---------------------------------------------------------------------------
__device__ __forceinline__ void gemm_copy_tile(const __bf16* __restrict__ A,
                                               const __bf16* __restrict__ BT,
                                               __bf16* As, __bf16* Bs,
                                               int m0, int n0, int k0, int K, int tid) {
  for (int c = tid; c < 512; c += 256) {
    int row = c >> 2, cc = c & 3;
    g2l16(A  + (size_t)(m0 + row) * K + k0 + cc * 8, (U128*)As + c);
    g2l16(BT + (size_t)(n0 + row) * K + k0 + cc * 8, (U128*)Bs + c);
  }
}

__global__ __launch_bounds__(256) void gemm128_kernel(const __bf16* __restrict__ A,
                                                      const __bf16* __restrict__ BT,
                                                      float* __restrict__ C,
                                                      int M, int N, int K) {
  __shared__ __align__(16) __bf16 As[2][128 * 32];
  __shared__ __align__(16) __bf16 Bs[2][128 * 32];

  const int tid  = threadIdx.x;
  const int wid  = tid >> 5;
  const int lane = tid & 31;
  const int lh   = lane >> 4;
  const int ln   = lane & 15;
  const int wm   = (wid & 3) * 32;
  const int wn   = (wid >> 2) * 64;
  const int m0   = blockIdx.y * 128;
  const int n0   = blockIdx.x * 128;

  v8f acc[2][4] = {};

  gemm_copy_tile(A, BT, As[0], Bs[0], m0, n0, 0, K, tid);
  wait_async();
  __syncthreads();

  int buf = 0;
  for (int k0 = 0; k0 < K; k0 += 32) {
    const int nxt = buf ^ 1;
    if (k0 + 32 < K)
      gemm_copy_tile(A, BT, As[nxt], Bs[nxt], m0, n0, k0 + 32, K, tid);

    Frag a[2];
#pragma unroll
    for (int i = 0; i < 2; ++i) {
      const __bf16* ar = As[buf] + (wm + i * 16 + ln) * 32;
      a[i].q[0] = *(const U128*)(ar + lh * 8);
      a[i].q[1] = *(const U128*)(ar + 16 + lh * 8);
    }
#pragma unroll
    for (int nt = 0; nt < 4; ++nt) {
      Frag b;
      const __bf16* br = Bs[buf] + (wn + nt * 16 + ln) * 32;
      b.q[0] = *(const U128*)(br + lh * 8);
      b.q[1] = *(const U128*)(br + 16 + lh * 8);
#pragma unroll
      for (int i = 0; i < 2; ++i)
        acc[i][nt] = __builtin_amdgcn_wmma_f32_16x16x32_bf16(
            false, a[i].v, false, b.v, (short)0, acc[i][nt], false, false);
    }
    wait_async();
    __syncthreads();
    buf = nxt;
  }

#pragma unroll
  for (int i = 0; i < 2; ++i)
#pragma unroll
    for (int nt = 0; nt < 4; ++nt)
#pragma unroll
      for (int r = 0; r < 8; ++r)
        C[(size_t)(m0 + wm + i * 16 + lh * 8 + r) * N + n0 + wn + nt * 16 + ln] =
            acc[i][nt][r];
}

// ---------------------------------------------------------------------------
// Small GEMM for N==64 (k_rope path): tile 128(M) x 64(N), 8 waves x 16 rows.
// ---------------------------------------------------------------------------
__global__ __launch_bounds__(256) void gemm64n_kernel(const __bf16* __restrict__ A,
                                                      const __bf16* __restrict__ BT,
                                                      float* __restrict__ C, int K) {
  const int N = 64;
  __shared__ __align__(16) __bf16 As[128 * 32];
  __shared__ __align__(16) __bf16 Bs[64 * 32];

  const int tid  = threadIdx.x;
  const int wid  = tid >> 5;
  const int lane = tid & 31;
  const int lh   = lane >> 4;
  const int ln   = lane & 15;
  const int m0   = blockIdx.y * 128;

  v8f acc[4] = {};

  for (int k0 = 0; k0 < K; k0 += 32) {
    __syncthreads();
    for (int c = tid; c < 512; c += 256) {
      int row = c >> 2, cc = c & 3;
      g2l16(A + (size_t)(m0 + row) * K + k0 + cc * 8, (U128*)As + c);
    }
    {
      int c = tid;
      if (c < 256) {
        int row = c >> 2, cc = c & 3;
        g2l16(BT + (size_t)row * K + k0 + cc * 8, (U128*)Bs + c);
      }
    }
    wait_async();
    __syncthreads();

    Frag a;
    const __bf16* ar = As + (wid * 16 + ln) * 32;
    a.q[0] = *(const U128*)(ar + lh * 8);
    a.q[1] = *(const U128*)(ar + 16 + lh * 8);
#pragma unroll
    for (int nt = 0; nt < 4; ++nt) {
      Frag b;
      const __bf16* br = Bs + (nt * 16 + ln) * 32;
      b.q[0] = *(const U128*)(br + lh * 8);
      b.q[1] = *(const U128*)(br + 16 + lh * 8);
      acc[nt] = __builtin_amdgcn_wmma_f32_16x16x32_bf16(
          false, a.v, false, b.v, (short)0, acc[nt], false, false);
    }
  }

#pragma unroll
  for (int nt = 0; nt < 4; ++nt)
#pragma unroll
    for (int r = 0; r < 8; ++r)
      C[(size_t)(m0 + wid * 16 + lh * 8 + r) * N + nt * 16 + ln] = acc[nt][r];
}

// ---------------------------------------------------------------------------
// Row-wise RMSNorm: out = scale * x / (rms + eps), rms = sqrt(sum(x^2))/sqrt(D)
// ---------------------------------------------------------------------------
__global__ __launch_bounds__(256) void rmsnorm_kernel(const float* __restrict__ in,
                                                      const float* __restrict__ scale,
                                                      __bf16* __restrict__ out, int D) {
  __shared__ float red[256];
  const size_t row = blockIdx.x;
  const float* p = in + row * (size_t)D;
  float s = 0.f;
  for (int i = threadIdx.x; i < D; i += 256) { float v = p[i]; s += v * v; }
  red[threadIdx.x] = s;
  __syncthreads();
  for (int off = 128; off > 0; off >>= 1) {
    if (threadIdx.x < off) red[threadIdx.x] += red[threadIdx.x + off];
    __syncthreads();
  }
  float rms = sqrtf(red[0]) * rsqrtf((float)D);
  float inv = 1.f / (rms + 1e-6f);
  __bf16* o = out + row * (size_t)D;
  for (int i = threadIdx.x; i < D; i += 256) o[i] = f2bf(p[i] * inv * scale[i]);
}

// ---------------------------------------------------------------------------
// Pack q: q_nope(b,s,h,64) f32 + rope(q_rope(b,s,h,64)) -> q(b,h,s,128) bf16
// ---------------------------------------------------------------------------
__global__ __launch_bounds__(64) void pack_q_kernel(const float* __restrict__ qn,
                                                    const float* __restrict__ qr,
                                                    const float* __restrict__ cosT,
                                                    const float* __restrict__ sinT,
                                                    __bf16* __restrict__ q) {
  int idx = blockIdx.x;
  int s = idx & (SC - 1);
  int h = (idx >> 11) & (HC - 1);
  int b = idx >> 15;
  size_t src = ((size_t)(b * SC + s) * HC + h) * 64;
  size_t dst = (size_t)idx * 128;
  int d = threadIdx.x;
  q[dst + d] = f2bf(qn[src + d]);
  if (d < 32) {
    float xr = qr[src + 2 * d], xi = qr[src + 2 * d + 1];
    float c = cosT[s * 32 + d], sn = sinT[s * 32 + d];
    q[dst + 64 + 2 * d]     = f2bf(xr * c - xi * sn);
    q[dst + 64 + 2 * d + 1] = f2bf(xr * sn + xi * c);
  }
}

// ---------------------------------------------------------------------------
// Pack k: k_nope(b,s,h,64) f32 + rope(k_rope(b,s,64)/H) -> k(b,h,s,128) bf16
// ---------------------------------------------------------------------------
__global__ __launch_bounds__(64) void pack_k_kernel(const float* __restrict__ kn,
                                                    const float* __restrict__ kr,
                                                    const float* __restrict__ cosT,
                                                    const float* __restrict__ sinT,
                                                    __bf16* __restrict__ k) {
  int idx = blockIdx.x;
  int s = idx & (SC - 1);
  int h = (idx >> 11) & (HC - 1);
  int b = idx >> 15;
  size_t src = ((size_t)(b * SC + s) * HC + h) * 64;
  size_t dst = (size_t)idx * 128;
  int d = threadIdx.x;
  k[dst + d] = f2bf(kn[src + d]);
  if (d < 32) {
    const float invH = 1.f / (float)HC;
    size_t rsrc = (size_t)(b * SC + s) * 64;
    float xr = kr[rsrc + 2 * d] * invH, xi = kr[rsrc + 2 * d + 1] * invH;
    float c = cosT[s * 32 + d], sn = sinT[s * 32 + d];
    k[dst + 64 + 2 * d]     = f2bf(xr * c - xi * sn);
    k[dst + 64 + 2 * d + 1] = f2bf(xr * sn + xi * c);
  }
}

// v(b,s,h,64) f32 -> vt(b,h,64,s) bf16  (pre-transposed for the P@V WMMA)
__global__ __launch_bounds__(64) void pack_v_kernel(const float* __restrict__ vf,
                                                    __bf16* __restrict__ vt) {
  int idx = blockIdx.x;   // b*H*S + h*S + s
  int s = idx & (SC - 1);
  int h = (idx >> 11) & (HC - 1);
  int b = idx >> 15;
  int bh = idx >> 11;
  int d = threadIdx.x;
  vt[((size_t)bh * 64 + d) * SC + s] =
      f2bf(vf[((size_t)(b * SC + s) * HC + h) * 64 + d]);
}

// ---------------------------------------------------------------------------
// Flash attention: per (b,h), 64 q-rows per block (4 waves x 16 rows).
// q,k: (B,H,S,128) bf16 ; vt: (B,H,64,S) bf16 ; out: (B,S,H*64) bf16
// ---------------------------------------------------------------------------
__global__ __launch_bounds__(128) void attn_kernel(const __bf16* __restrict__ q,
                                                   const __bf16* __restrict__ k,
                                                   const __bf16* __restrict__ vt,
                                                   __bf16* __restrict__ out) {
  __shared__ __align__(16) __bf16 kT[64 * 128];      // [krow][d]
  __shared__ __align__(16) __bf16 vT[64 * 64];       // [vd][krow]
  __shared__ __align__(16) __bf16 pT[4][16 * 64];    // per-wave P tile [m][krow]

  const int tid  = threadIdx.x;
  const int wave = tid >> 5;
  const int lane = tid & 31;
  const int lh   = lane >> 4;
  const int ln   = lane & 15;
  const int qt   = blockIdx.x;
  const int bh   = blockIdx.y;
  const int b    = bh >> 4;
  const int h    = bh & 15;
  const size_t qkbase = (size_t)bh * SC * 128;
  const __bf16* vbase = vt + (size_t)bh * 64 * SC;
  const int q0 = qt * 64;

  // Preload this wave's 16 q rows as 4 K-chunk A-fragments
  Frag qf[4];
  {
    const __bf16* qrow = q + qkbase + (size_t)(q0 + wave * 16 + ln) * 128;
#pragma unroll
    for (int kc = 0; kc < 4; ++kc) {
      qf[kc].q[0] = *(const U128*)(qrow + kc * 32 + lh * 8);
      qf[kc].q[1] = *(const U128*)(qrow + kc * 32 + 16 + lh * 8);
    }
  }

  v8f oacc[4] = {};
  float mrow[8], lrow[8];
#pragma unroll
  for (int r = 0; r < 8; ++r) { mrow[r] = -3.0e38f; lrow[r] = 0.f; }
  const float sscale = 0.08838834764831845f;  // 1/sqrt(128)

  for (int kt = 0; kt <= qt; ++kt) {
    __syncthreads();
    // Stage k tile (64x128) and v tile ([vd][krow]) -- all b128 copies
    {
      const __bf16* ksrc = k + qkbase + (size_t)kt * 64 * 128;
      for (int c = tid; c < 1024; c += 128)
        g2l16(ksrc + (size_t)c * 8, (U128*)kT + c);
      const __bf16* vsrc = vbase + kt * 64;
      for (int c = tid; c < 512; c += 128) {
        int row = c >> 3, cc = c & 7;
        g2l16(vsrc + (size_t)row * SC + cc * 8, (U128*)vT + c);
      }
    }
    wait_async();
    __syncthreads();

    // S = q @ k^T  (16 x 64 per wave)
    v8f sacc[4] = {};
#pragma unroll
    for (int kc = 0; kc < 4; ++kc) {
#pragma unroll
      for (int nt = 0; nt < 4; ++nt) {
        Frag bfr;
        const __bf16* kr = kT + (nt * 16 + ln) * 128 + kc * 32;
        bfr.q[0] = *(const U128*)(kr + lh * 8);
        bfr.q[1] = *(const U128*)(kr + 16 + lh * 8);
        sacc[nt] = __builtin_amdgcn_wmma_f32_16x16x32_bf16(
            false, qf[kc].v, false, bfr.v, (short)0, sacc[nt], false, false);
      }
    }

    // Online softmax per row (row = lh*8 + r; 16 lanes of the half hold it)
    float pv[4][8];
#pragma unroll
    for (int r = 0; r < 8; ++r) {
      int qg = q0 + wave * 16 + lh * 8 + r;
      float rm = -3.0e38f;
#pragma unroll
      for (int nt = 0; nt < 4; ++nt) {
        int kg = kt * 64 + nt * 16 + ln;
        float sv = (kg <= qg) ? sacc[nt][r] * sscale : -3.0e38f;
        pv[nt][r] = sv;
        rm = fmaxf(rm, sv);
      }
      for (int off = 1; off < 16; off <<= 1) rm = fmaxf(rm, __shfl_xor(rm, off, 32));
      float mnew = fmaxf(mrow[r], rm);
      float corr = __expf(mrow[r] - mnew);
      float rs = 0.f;
#pragma unroll
      for (int nt = 0; nt < 4; ++nt) {
        float e = __expf(pv[nt][r] - mnew);
        pv[nt][r] = e;
        rs += e;
      }
      for (int off = 1; off < 16; off <<= 1) rs += __shfl_xor(rs, off, 32);
      lrow[r] = lrow[r] * corr + rs;
      mrow[r] = mnew;
#pragma unroll
      for (int nt = 0; nt < 4; ++nt) oacc[nt][r] *= corr;
    }

    // P -> LDS (bf16), then O += P @ V
#pragma unroll
    for (int nt = 0; nt < 4; ++nt)
#pragma unroll
      for (int r = 0; r < 8; ++r)
        pT[wave][(lh * 8 + r) * 64 + nt * 16 + ln] = f2bf(pv[nt][r]);
    __syncthreads();

#pragma unroll
    for (int kc = 0; kc < 2; ++kc) {
      Frag pf;
      const __bf16* pr = pT[wave] + ln * 64 + kc * 32;
      pf.q[0] = *(const U128*)(pr + lh * 8);
      pf.q[1] = *(const U128*)(pr + 16 + lh * 8);
#pragma unroll
      for (int nt = 0; nt < 4; ++nt) {
        Frag vf;
        const __bf16* vr = vT + (nt * 16 + ln) * 64 + kc * 32;
        vf.q[0] = *(const U128*)(vr + lh * 8);
        vf.q[1] = *(const U128*)(vr + 16 + lh * 8);
        oacc[nt] = __builtin_amdgcn_wmma_f32_16x16x32_bf16(
            false, pf.v, false, vf.v, (short)0, oacc[nt], false, false);
      }
    }
  }

  // Normalize and write (B,S,H*64) bf16
#pragma unroll
  for (int r = 0; r < 8; ++r) {
    int sg = q0 + wave * 16 + lh * 8 + r;
    float invl = 1.f / lrow[r];
    size_t obase = ((size_t)(b * SC + sg)) * (HC * 64) + h * 64;
#pragma unroll
    for (int nt = 0; nt < 4; ++nt)
      out[obase + nt * 16 + ln] = f2bf(oacc[nt][r] * invl);
  }
}

// ---------------------------------------------------------------------------
// Host orchestration
// ---------------------------------------------------------------------------
static inline void* bump(char* ws, size_t& off, size_t bytes) {
  off = (off + 255) & ~(size_t)255;
  void* p = ws + off;
  off += bytes;
  return p;
}

extern "C" void kernel_launch(void* const* d_in, const int* in_sizes, int n_in,
                              void* d_out, int out_size, void* d_ws, size_t ws_size,
                              hipStream_t stream) {
  const float* x       = (const float*)d_in[0];
  const float* Wcq     = (const float*)d_in[1];
  const float* q_scale = (const float*)d_in[2];
  const float* Wq_nope = (const float*)d_in[3];
  const float* Wq_rope = (const float*)d_in[4];
  const float* Wckv    = (const float*)d_in[5];
  const float* kv_scale= (const float*)d_in[6];
  const float* Wk_nope = (const float*)d_in[7];
  const float* Wv      = (const float*)d_in[8];
  const float* Wk_rope = (const float*)d_in[9];
  const float* Wproj   = (const float*)d_in[10];
  const float* cosT    = (const float*)d_in[11];
  const float* sinT    = (const float*)d_in[12];

  char* ws = (char*)d_ws;
  size_t off = 0;
  const int M = MROWS;             // 4096
  const int HD = HC * 64;          // 1024

  __bf16* xb      = (__bf16*)bump(ws, off, (size_t)M * DIMC * 2);
  __bf16* WcqT    = (__bf16*)bump(ws, off, (size_t)DIMC * QLORA * 2);   // [N][K]
  __bf16* WqnT    = (__bf16*)bump(ws, off, (size_t)QLORA * HD * 2);
  __bf16* WqrT    = (__bf16*)bump(ws, off, (size_t)QLORA * HD * 2);
  __bf16* WckvT   = (__bf16*)bump(ws, off, (size_t)DIMC * KVLORA * 2);
  __bf16* WknT    = (__bf16*)bump(ws, off, (size_t)KVLORA * HD * 2);
  __bf16* WvT     = (__bf16*)bump(ws, off, (size_t)KVLORA * HD * 2);
  __bf16* WkrT    = (__bf16*)bump(ws, off, (size_t)DIMC * 64 * 2);
  __bf16* WprojT  = (__bf16*)bump(ws, off, (size_t)HD * DIMC * 2);
  __bf16* nqb     = (__bf16*)bump(ws, off, (size_t)M * QLORA * 2);
  __bf16* nkvb    = (__bf16*)bump(ws, off, (size_t)M * KVLORA * 2);
  __bf16* qb      = (__bf16*)bump(ws, off, (size_t)M * HC * 128 * 2);   // (B,H,S,128)
  __bf16* kb      = (__bf16*)bump(ws, off, (size_t)M * HC * 128 * 2);
  __bf16* vtb     = (__bf16*)bump(ws, off, (size_t)M * HC * 64 * 2);    // (B,H,64,S)
  __bf16* attnb   = (__bf16*)bump(ws, off, (size_t)M * HD * 2);         // (B,S,H*64)
  float*  F1      = (float*)bump(ws, off, (size_t)M * QLORA * 4);
  float*  F2      = (float*)bump(ws, off, (size_t)M * HD * 4);
  float*  F3      = (float*)bump(ws, off, (size_t)M * HD * 4);

  // Conversions: activations row-major, weights transposed to [N][K]
  cvt_bf16_kernel<<<(M * DIMC + 255) / 256, 256, 0, stream>>>(x, xb, M * DIMC);
  auto cvtT = [&](const float* src, __bf16* dst, int K, int N) {
    cvtT_bf16_kernel<<<dim3(N / 32, K / 32), dim3(32, 8), 0, stream>>>(src, dst, K, N);
  };
  cvtT(Wcq,     WcqT,   DIMC, QLORA);
  cvtT(Wq_nope, WqnT,   QLORA, HD);
  cvtT(Wq_rope, WqrT,   QLORA, HD);
  cvtT(Wckv,    WckvT,  DIMC, KVLORA);
  cvtT(Wk_nope, WknT,   KVLORA, HD);
  cvtT(Wv,      WvT,    KVLORA, HD);
  cvtT(Wk_rope, WkrT,   DIMC, 64);
  cvtT(Wproj,   WprojT, HD, DIMC);

  auto gemm = [&](const __bf16* A, const __bf16* BT, float* C, int m, int n, int kk) {
    gemm128_kernel<<<dim3(n / 128, m / 128), 256, 0, stream>>>(A, BT, C, m, n, kk);
  };

  // Q path
  gemm(xb, WcqT, F1, M, QLORA, DIMC);
  rmsnorm_kernel<<<M, 256, 0, stream>>>(F1, q_scale, nqb, QLORA);
  gemm(nqb, WqnT, F2, M, HD, QLORA);
  gemm(nqb, WqrT, F3, M, HD, QLORA);
  pack_q_kernel<<<BC * HC * SC, 64, 0, stream>>>(F2, F3, cosT, sinT, qb);

  // KV path
  gemm(xb, WckvT, F1, M, KVLORA, DIMC);
  rmsnorm_kernel<<<M, 256, 0, stream>>>(F1, kv_scale, nkvb, KVLORA);
  gemm(nkvb, WknT, F2, M, HD, KVLORA);
  gemm64n_kernel<<<dim3(1, M / 128), 256, 0, stream>>>(xb, WkrT, F3, DIMC);
  pack_k_kernel<<<BC * HC * SC, 64, 0, stream>>>(F2, F3, cosT, sinT, kb);
  gemm(nkvb, WvT, F2, M, HD, KVLORA);
  pack_v_kernel<<<BC * HC * SC, 64, 0, stream>>>(F2, vtb);

  // Attention
  attn_kernel<<<dim3(SC / 64, BC * HC), 128, 0, stream>>>(qb, kb, vtb, attnb);

  // Output projection -> d_out (f32)
  gemm(attnb, WprojT, (float*)d_out, M, DIMC, HD);
}
